// WTLSTM_21492016350024
// MI455X (gfx1250) — matrix operand order repaired and checked by reading, not compiled, not run on hardware
//
#include <hip/hip_runtime.h>
#include <hip/hip_bf16.h>
#include <math.h>

// ---------------------------------------------------------------------------
// Fused WT + LSTM(3->64) + LSTM(64->32) + FC(11520->128) + FC(128->5) + softmax
// One workgroup (4 wave32) owns 16 batch rows; all sequence state stays in
// registers/LDS; matrix work on v_wmma_f32_16x16x32_f16; activations on native
// v_tanh_f32 / v_exp_f32 / v_rcp_f32; W1 pre-converted to f16 in workspace.
// ---------------------------------------------------------------------------

typedef _Float16 h16;
typedef __attribute__((ext_vector_type(16))) _Float16 v16h;
typedef __attribute__((ext_vector_type(8)))  _Float16 v8h;
typedef __attribute__((ext_vector_type(8)))  float    v8f;
typedef __attribute__((ext_vector_type(4)))  int      v4i32;

#define WMMA_F16(a, b, c) \
  __builtin_amdgcn_wmma_f32_16x16x32_f16(false, (a), false, (b), (short)0, (c), false, false)

#if defined(__gfx1250__) && defined(__has_builtin)
#if __has_builtin(__builtin_amdgcn_global_load_async_to_lds_b128)
#define HAVE_ASYNC_LDS 1
#endif
#endif
#ifndef HAVE_ASYNC_LDS
#define HAVE_ASYNC_LDS 0
#endif

// Fast activations on native transcendentals (v_exp_f32 / v_rcp_f32 / v_tanh_f32)
__device__ __forceinline__ float sigf(float x) {
  return __builtin_amdgcn_rcpf(1.0f + __builtin_amdgcn_exp2f(-1.44269504089f * x));
}
#if defined(__gfx1250__) && defined(__has_builtin)
#if __has_builtin(__builtin_amdgcn_tanhf)
#define FAST_TANH 1
#endif
#endif
__device__ __forceinline__ float tanh_(float x) {
#if defined(FAST_TANH)
  return __builtin_amdgcn_tanhf(x);
#else
  // tanh(x) = 2*sigmoid(2x) - 1
  return __builtin_fmaf(2.0f, sigf(2.0f * x), -1.0f);
#endif
}

// pywt 'symmetric' DWT filters (DEC_LO = REC_LO reversed, DEC_HI = QMF)
__device__ __constant__ float DEC_LO_C[12] = {
  -0.00107730108499558f,   0.004777257511010651f,  0.0005538422009938016f,
  -0.031582039318031156f,  0.02752286553001629f,   0.09750160558707936f,
  -0.12976686756709563f,  -0.22626469396516913f,   0.3152503517092432f,
   0.7511339080215775f,    0.4946238903983854f,    0.11154074335008017f };
__device__ __constant__ float DEC_HI_C[12] = {
  -0.11154074335008017f,   0.4946238903983854f,   -0.7511339080215775f,
   0.3152503517092432f,    0.22626469396516913f,  -0.12976686756709563f,
  -0.09750160558707936f,   0.02752286553001629f,   0.031582039318031156f,
   0.0005538422009938016f, -0.004777257511010651f, -0.00107730108499558f };

__device__ __forceinline__ int symi(int i, int n) {
  if (i < 0)  i = -i - 1;
  if (i >= n) i = 2 * n - 1 - i;
  return i;
}

// One-time W1 f32 -> f16 conversion into workspace
__global__ __launch_bounds__(256)
void cvt_w1_f16(const float* __restrict__ W1, h16* __restrict__ W1h, int ntot) {
  int i = blockIdx.x * 256 + threadIdx.x;
  if (i < ntot) W1h[i] = (h16)W1[i];
}

__global__ __launch_bounds__(128)
void wtlstm_fused(const float* __restrict__ x,
                  const float* __restrict__ Wih1, const float* __restrict__ Whh1,
                  const float* __restrict__ bih1, const float* __restrict__ bhh1,
                  const float* __restrict__ Wih2, const float* __restrict__ Whh2,
                  const float* __restrict__ bih2, const float* __restrict__ bhh2,
                  const float* __restrict__ W1,  const h16* __restrict__ W1h,
                  const float* __restrict__ b1,
                  const float* __restrict__ W2,  const float* __restrict__ b2,
                  float* __restrict__ out)
{
  constexpr int Tn = 360;        // sequence length
  constexpr int KF = 11520;      // FC1 reduction dim (= 360*32)

  __shared__ __align__(16) h16   ssig[Tn][16];       // signal feature (f16)
  __shared__ __align__(16) h16   d2d1[Tn][16][2];    // [t][row][{cD2,cD1}], zero-padded
  __shared__ __align__(16) float ca1s[16][192];      // cA1 intermediate
  __shared__ __align__(16) h16   hbuf1[16][64];      // LSTM1 hidden (f16)
  __shared__ __align__(16) float gbuf2[16][128];     // LSTM2 gate exchange / z-buffer
  __shared__ __align__(16) h16   h2buf[16][32];      // LSTM2 hidden (f16)
  __shared__ __align__(16) float w2s[5 * 128];       // staged W2
  __shared__ __align__(16) float zrow[16][5];        // logits

  const int tid  = threadIdx.x;
  const int lane = tid & 31;
  const int w    = tid >> 5;           // wave id 0..3
  const int blk  = blockIdx.x;
  const int row0 = blk * 16;
  const float* xb = x + (size_t)row0 * Tn;

  const int  n     = lane & 15;               // B-frag column / C-frag column
  const int  m     = lane & 15;               // A-frag row
  const bool lo16  = (lane < 16);
  const int  koff  = lo16 ? 0 : 16;           // B-frag K half
  const int  badd  = lo16 ? 0 : 8;            // C-frag row offset
  const int  abase = lo16 ? 0 : 8;            // unused (A-frags use koff)
  const int  hbase = 16 * w;

  // ---------------- Phase A: stage W2 (async) + wavelet features -----------
#if HAVE_ASYNC_LDS
  {
    typedef __attribute__((address_space(1))) v4i32* gv4_t;
    typedef __attribute__((address_space(3))) v4i32* lv4_t;
    unsigned long long gsrc  = (unsigned long long)(const void*)W2;
    unsigned           lbase = (unsigned)(unsigned long long)(const void*)&w2s[0];
    for (int i = tid; i < 160; i += 128)   // 5*128*4 bytes = 160 x b128
      __builtin_amdgcn_global_load_async_to_lds_b128(
          (gv4_t)(gsrc + 16ull * i), (lv4_t)(unsigned long long)(lbase + 16u * i), 0, 0);
  }
#else
  for (int i = tid; i < 640; i += 128) w2s[i] = W2[i];
#endif

  for (int i = tid; i < 16 * Tn; i += 128) {
    int r = i / Tn, t = i - r * Tn;
    ssig[t][r] = (h16)xb[r * Tn + t];
  }
  for (int i = tid; i < Tn * 16; i += 128) {
    int t = i >> 4, r = i & 15;
    d2d1[t][r][0] = (h16)0.f;
    d2d1[t][r][1] = (h16)0.f;
  }
  __syncthreads();

  // cA1 / cD1: y[j] = sum_u filt[u] * s[sym(2j+1-u)]
  for (int i = tid; i < 16 * 185; i += 128) {
    int r = i / 185, j = i - r * 185;
    float lo = 0.f, hi = 0.f;
#pragma unroll
    for (int u = 0; u < 12; ++u) {
      float v = xb[r * Tn + symi(2 * j + 1 - u, Tn)];
      lo += DEC_LO_C[u] * v;
      hi += DEC_HI_C[u] * v;
    }
    ca1s[r][j]    = lo;
    d2d1[j][r][1] = (h16)hi;        // cD1
  }
  __syncthreads();
  for (int i = tid; i < 16 * 98; i += 128) {
    int r = i / 98, j = i - r * 98;
    float hi = 0.f;
#pragma unroll
    for (int u = 0; u < 12; ++u)
      hi += DEC_HI_C[u] * ca1s[r][symi(2 * j + 1 - u, 185)];
    d2d1[j][r][0] = (h16)hi;        // cD2
  }

  // ---------------- Phase B: resident weight fragments ----------------------
  // LSTM1: wave w owns hidden units [16w,16w+16) -> 4 gate tiles, K=64.
  // Bias folded into spare K=3 slot of the x-projection (A has constant 1.0).
  v16h whh1f[4][2];
  v16h wih1f[4];
#pragma unroll
  for (int g = 0; g < 4; ++g) {
    int col = g * 64 + hbase + n;
#pragma unroll
    for (int kc = 0; kc < 2; ++kc)
#pragma unroll
      for (int e = 0; e < 16; ++e)
        whh1f[g][kc][e] = (h16)Whh1[col * 64 + kc * 32 + koff + e];
#pragma unroll
    for (int e = 0; e < 16; ++e) {
      int k = koff + e;
      h16 v = (h16)0.f;
      if (k < 3)       v = (h16)Wih1[col * 3 + k];
      else if (k == 3) v = (h16)(bih1[col] + bhh1[col]);   // bias via 1.0 feature
      wih1f[g][e] = v;
    }
  }
  // LSTM2: wave w owns gate w for hidden tiles 0,1 (cols w*32+ht*16+n)
  v16h wih2f[2][2];
  v16h whh2f[2];
#pragma unroll
  for (int ht = 0; ht < 2; ++ht) {
    int col = w * 32 + ht * 16 + n;
#pragma unroll
    for (int kc = 0; kc < 2; ++kc)
#pragma unroll
      for (int e = 0; e < 16; ++e)
        wih2f[ht][kc][e] = (h16)Wih2[col * 64 + kc * 32 + koff + e];
#pragma unroll
    for (int e = 0; e < 16; ++e)
      whh2f[ht][e] = (h16)Whh2[col * 32 + koff + e];
  }
  // LSTM2 per-column gate biases, applied in the activation phase (waves 0,1)
  float b2r[4] = {0.f, 0.f, 0.f, 0.f};
  if (w < 2) {
    int hid = w * 16 + n;
#pragma unroll
    for (int g = 0; g < 4; ++g)
      b2r[g] = bih2[g * 32 + hid] + bhh2[g * 32 + hid];
  }

  v8f c1 = {}, c2 = {};
  v16h ahA = {}, ahB = {}, a2h = {};
  v8f zacc0 = {}, zacc1 = {};
  const h16 hz = (h16)0.f;
  __syncthreads();

  // ---------------- Phase C: fused recurrence over T=360 --------------------
  for (int t = 0; t < Tn; ++t) {
    // feats A-fragment: branchless, 3 LDS loads + selects; K=3 carries 1.0
    v16h af = {};
    {
      h16 f0 = ssig[t][m];
      h16 f1 = d2d1[t][m][0];
      h16 f2 = d2d1[t][m][1];
      af[0] = lo16 ? f0 : hz;
      af[1] = lo16 ? f1 : hz;
      af[2] = lo16 ? f2 : hz;
      af[3] = lo16 ? (h16)1.0f : hz;
    }

    // ---- LSTM1 gates: (x|1)@ (Wih|b)T + h@WhhT, C starts at 0 (12 WMMAs) ----
    v8f acc[4];
#pragma unroll
    for (int g = 0; g < 4; ++g) {
      v8f a = {};
      a = WMMA_F16(af,  wih1f[g],    a);
      a = WMMA_F16(ahA, whh1f[g][0], a);
      a = WMMA_F16(ahB, whh1f[g][1], a);
      acc[g] = a;
    }
#pragma unroll
    for (int vv = 0; vv < 8; ++vv) {
      float ig = sigf(acc[0][vv]);
      float fg = sigf(acc[1][vv]);
      float gg = tanh_(acc[2][vv]);
      float og = sigf(acc[3][vv]);
      float cc = fg * c1[vv] + ig * gg;
      c1[vv] = cc;
      hbuf1[vv + badd][hbase + n] = (h16)(og * tanh_(cc));
    }
    __syncthreads();

    // rebuild h1 A-fragments: K-contiguous halves -> 2x ds_load_b128 each
    {
      const h16* hp = &hbuf1[m][0];
      v8h a0 = *(const v8h*)(hp + koff);             // K = koff..koff+7
      v8h a1 = *(const v8h*)(hp + 8 + koff);         // K = 8+koff..
      v8h b0 = *(const v8h*)(hp + 32 + koff);        // K = 32+koff..
      v8h b1v = *(const v8h*)(hp + 40 + koff);       // K = 40+koff..
      ahA = __builtin_shufflevector(a0, a1, 0,1,2,3,4,5,6,7, 8,9,10,11,12,13,14,15);
      ahB = __builtin_shufflevector(b0, b1v, 0,1,2,3,4,5,6,7, 8,9,10,11,12,13,14,15);
    }

    // ---- LSTM2 gates: wave w computes gate w, C starts at 0 (6 WMMAs) ------
#pragma unroll
    for (int ht = 0; ht < 2; ++ht) {
      v8f a = {};
      a = WMMA_F16(ahA, wih2f[ht][0], a);
      a = WMMA_F16(ahB, wih2f[ht][1], a);
      a = WMMA_F16(a2h, whh2f[ht],    a);
#pragma unroll
      for (int vv = 0; vv < 8; ++vv)
        gbuf2[vv + badd][w * 32 + ht * 16 + n] = a[vv];
    }
    __syncthreads();

    if (w < 2) {  // waves 0,1 update c2/h2 for hidden tiles 0,1 (bias here)
#pragma unroll
      for (int vv = 0; vv < 8; ++vv) {
        int M = vv + badd, hid = w * 16 + n;
        float ig = sigf(gbuf2[M][hid]       + b2r[0]);
        float fg = sigf(gbuf2[M][32 + hid]  + b2r[1]);
        float gg = tanh_(gbuf2[M][64 + hid] + b2r[2]);
        float og = sigf(gbuf2[M][96 + hid]  + b2r[3]);
        float cc = fg * c2[vv] + ig * gg;
        c2[vv] = cc;
        h2buf[M][hid] = (h16)(og * tanh_(cc));
      }
    }
    __syncthreads();

    // h2 A-fragment: 2x ds_load_b128
    {
      const h16* hp = &h2buf[m][0];
      v8h a0 = *(const v8h*)(hp + koff);             // K = koff..koff+7
      v8h a1 = *(const v8h*)(hp + 8 + koff);         // K = 8+koff..
      a2h = __builtin_shufflevector(a0, a1, 0,1,2,3,4,5,6,7, 8,9,10,11,12,13,14,15);
    }

    // ---- FC1 on-the-fly accumulation: z += h2_t @ W1_chunk_t^T (2 WMMAs) ----
    if (W1h) {     // f16 path: 4x global_load_b128, no conversions
      const h16* wr0 = W1h + (size_t)(w * 32 + n)      * KF + t * 32 + koff;
      const h16* wr1 = W1h + (size_t)(w * 32 + 16 + n) * KF + t * 32 + koff;
      v8h p0 = *(const v8h*)wr0;
      v8h p1 = *(const v8h*)(wr0 + 8);
      v8h p2 = *(const v8h*)wr1;
      v8h p3 = *(const v8h*)(wr1 + 8);
      v16h bw0 = __builtin_shufflevector(p0, p1, 0,1,2,3,4,5,6,7, 8,9,10,11,12,13,14,15);
      v16h bw1 = __builtin_shufflevector(p2, p3, 0,1,2,3,4,5,6,7, 8,9,10,11,12,13,14,15);
      zacc0 = WMMA_F16(a2h, bw0, zacc0);
      zacc1 = WMMA_F16(a2h, bw1, zacc1);
      if (t + 1 < Tn) {                       // global_prefetch_b8 of next chunk
        __builtin_prefetch(wr0 + 32, 0, 1);
        __builtin_prefetch(wr1 + 32, 0, 1);
      }
    } else {       // fallback: convert f32 in-loop
      const float* wr0 = W1 + (size_t)(w * 32 + n)      * KF + t * 32 + koff;
      const float* wr1 = W1 + (size_t)(w * 32 + 16 + n) * KF + t * 32 + koff;
      v16h bw0, bw1;
      const float4* q0 = (const float4*)wr0;
      const float4* q1 = (const float4*)wr1;
#pragma unroll
      for (int p = 0; p < 4; ++p) {
        float4 a0 = q0[p], a1 = q1[p];
        bw0[4 * p + 0] = (h16)a0.x; bw0[4 * p + 1] = (h16)a0.y;
        bw0[4 * p + 2] = (h16)a0.z; bw0[4 * p + 3] = (h16)a0.w;
        bw1[4 * p + 0] = (h16)a1.x; bw1[4 * p + 1] = (h16)a1.y;
        bw1[4 * p + 2] = (h16)a1.z; bw1[4 * p + 3] = (h16)a1.w;
      }
      zacc0 = WMMA_F16(a2h, bw0, zacc0);
      zacc1 = WMMA_F16(a2h, bw1, zacc1);
      if (t + 1 < Tn) {
        __builtin_prefetch(wr0 + 32, 0, 1);
        __builtin_prefetch(wr1 + 32, 0, 1);
      }
    }
  }

  // ---------------- Phase D: relu + FC2 + softmax ---------------------------
#pragma unroll
  for (int vv = 0; vv < 8; ++vv) {
    int M = vv + badd;
    int col0 = w * 32 + n, col1 = w * 32 + 16 + n;
    gbuf2[M][col0] = fmaxf(zacc0[vv] + b1[col0], 0.f);
    gbuf2[M][col1] = fmaxf(zacc1[vv] + b1[col1], 0.f);
  }
#if HAVE_ASYNC_LDS
#if __has_builtin(__builtin_amdgcn_s_wait_asynccnt)
  __builtin_amdgcn_s_wait_asynccnt(0);
#else
  asm volatile("s_wait_asynccnt 0" ::: "memory");
#endif
#endif
  __syncthreads();

  if (tid < 80) {                    // 16 rows x 5 classes
    int r = tid / 5, q = tid - r * 5;
    float s = b2[q];
#pragma unroll 8
    for (int k = 0; k < 128; ++k) s += gbuf2[r][k] * w2s[q * 128 + k];
    zrow[r][q] = s;
  }
  __syncthreads();
  if (tid < 16) {
    int r = tid;
    float mx = zrow[r][0];
#pragma unroll
    for (int q = 1; q < 5; ++q) mx = fmaxf(mx, zrow[r][q]);
    float e[5], sum = 0.f;
#pragma unroll
    for (int q = 0; q < 5; ++q) {
      e[q] = __builtin_amdgcn_exp2f(1.44269504089f * (zrow[r][q] - mx));
      sum += e[q];
    }
    float inv = 1.f / sum;
#pragma unroll
    for (int q = 0; q < 5; ++q) out[(size_t)(row0 + r) * 5 + q] = e[q] * inv;
  }
}

extern "C" void kernel_launch(void* const* d_in, const int* in_sizes, int n_in,
                              void* d_out, int out_size, void* d_ws, size_t ws_size,
                              hipStream_t stream) {
  (void)n_in; (void)out_size;
  const float* x    = (const float*)d_in[0];
  const float* Wih1 = (const float*)d_in[1];
  const float* Whh1 = (const float*)d_in[2];
  const float* bih1 = (const float*)d_in[3];
  const float* bhh1 = (const float*)d_in[4];
  const float* Wih2 = (const float*)d_in[5];
  const float* Whh2 = (const float*)d_in[6];
  const float* bih2 = (const float*)d_in[7];
  const float* bhh2 = (const float*)d_in[8];
  const float* W1   = (const float*)d_in[9];
  const float* b1   = (const float*)d_in[10];
  const float* W2   = (const float*)d_in[11];
  const float* b2   = (const float*)d_in[12];

  const int n_w1 = 128 * 11520;
  h16* W1h = nullptr;
  if (ws_size >= (size_t)n_w1 * sizeof(h16)) {   // deterministic: same every call
    W1h = (h16*)d_ws;
    cvt_w1_f16<<<(n_w1 + 255) / 256, 256, 0, stream>>>(W1, W1h, n_w1);
  }

  const int B    = in_sizes[0] / 360;   // 4096
  const int grid = B / 16;              // 256 workgroups, 4 wave32 each
  wtlstm_fused<<<grid, 128, 0, stream>>>(x, Wih1, Whh1, bih1, bhh1,
                                         Wih2, Whh2, bih2, bhh2,
                                         W1, W1h, b1, W2, b2, (float*)d_out);
}
